// ReformerAttention_3040836845882
// MI455X (gfx1250) — compile-verified, matrix operand order checked by reading, not run on hardware
//
#include <hip/hip_runtime.h>
#include <hip/hip_bf16.h>
#include <math.h>

// ---- problem constants (fixed by the reference) ----
#define BB     4
#define TT     4096
#define HIDN   768
#define HH     12
#define DD     64
#define NHASH  2
#define NBUCK  32
#define CBUCK  (NHASH * NBUCK)   // 64 combined buckets
#define CL     128               // chunk length (bucket size)
#define NCHUNK 64                // n_hashes * n_buckets
#define BHN    (BB * HH)         // 48
#define TOTALN (NHASH * TT)      // 8192
#define MTOT   (BB * TT)         // 16384

typedef __attribute__((ext_vector_type(16))) __bf16 v16bf;
typedef __attribute__((ext_vector_type(8)))  float  v8f;

// ---- gfx1250 async global->LDS path (guarded: falls back to VGPR staging) ----
#if defined(__has_builtin)
#  if __has_builtin(__builtin_amdgcn_global_load_async_to_lds_b128) && \
      __has_builtin(__builtin_amdgcn_s_wait_asynccnt)
#    define HAVE_ASYNC_LDS 1
#  endif
#endif
#ifndef HAVE_ASYNC_LDS
#  define HAVE_ASYNC_LDS 0
#endif

#if HAVE_ASYNC_LDS
// Copy 32B (16 bf16) from global to LDS without touching VGPR data paths.
// Builtin signature (from clang diagnostics): (int4 AS1*, int4 AS3*, Imm, Imm).
// AS(1) from flat: numeric identity. AS(3) from flat: low 32 bits are the LDS
// address (aperture encoding, ISA 10.2), so an integer round-trip is exact.
typedef int v4i_g __attribute__((vector_size(4 * sizeof(int))));
typedef __attribute__((address_space(1))) v4i_g* as1_v4i;
typedef __attribute__((address_space(3))) v4i_g* as3_v4i;

static __device__ __forceinline__ void async_cp32(const __bf16* g, __bf16* l) {
    as1_v4i gp = (as1_v4i)(unsigned long long)(__SIZE_TYPE__)g;
    as3_v4i lp = (as3_v4i)(unsigned)(__SIZE_TYPE__)l;
    __builtin_amdgcn_global_load_async_to_lds_b128(gp, lp, 0, 0);
    __builtin_amdgcn_global_load_async_to_lds_b128(gp, lp, 16, 0);
}
#endif

// ---------------------------------------------------------------------------
// WMMA helpers (CDNA5 v_wmma_f32_16x16x32_bf16, layouts per ISA 7.12.2)
// ---------------------------------------------------------------------------
static __device__ __forceinline__ v8f wmma_bf16(v16bf a, v16bf b, v8f c) {
    return __builtin_amdgcn_wmma_f32_16x16x32_bf16(
        /*neg_a=*/false, a, /*neg_b=*/false, b,
        /*c_mod=*/(short)0, c, /*reuse_a=*/false, /*reuse_b=*/false);
}

// Load a 16x32 bf16 fragment from a row-major tile (LDS or global bf16).
// Lane L handles row (L&15); VGPR v holds K = (v>=4?16:0) + (L>=16?8:0) + (v&3)*2 {,+1}.
// Serves both the A operand (rows = M) and the B operand (rows = N of B^T).
static __device__ __forceinline__ v16bf frag_ld(const __bf16* base, int stride) {
    const int lane = threadIdx.x & 31;
    const __bf16* p = base + (lane & 15) * stride + (lane >> 4) * 8;
    v16bf f;
#pragma unroll
    for (int v = 0; v < 8; ++v) {
        const int k = ((v >> 2) << 4) + ((v & 3) << 1);
        f[2 * v]     = p[k];
        f[2 * v + 1] = p[k + 1];
    }
    return f;
}

// Same fragment pattern but from a per-lane f32 row pointer (row chosen by caller).
static __device__ __forceinline__ v16bf frag_row_f32(const float* rowp) {
    const int lane = threadIdx.x & 31;
    const float* p = rowp + (lane >> 4) * 8;
    v16bf f;
#pragma unroll
    for (int v = 0; v < 8; ++v) {
        const int k = ((v >> 2) << 4) + ((v & 3) << 1);
        f[2 * v]     = (__bf16)p[k];
        f[2 * v + 1] = (__bf16)p[k + 1];
    }
    return f;
}

// ---------------------------------------------------------------------------
// f32 -> bf16 elementwise convert
// ---------------------------------------------------------------------------
__global__ __launch_bounds__(256) void f32_to_bf16_kernel(
    const float* __restrict__ src, __bf16* __restrict__ dst, size_t n) {
    size_t i = (size_t)blockIdx.x * 256 + threadIdx.x;
    if (i < n) dst[i] = (__bf16)src[i];
}

// ---------------------------------------------------------------------------
// GEMM: C[M,N] = A[M,K](bf16) @ W[N,K](bf16)^T (+bias), C f32.
// Block: 256 thr = 8 waves (4 x 2), tile 128x64, K staged 32-wide in LDS via
// async global->LDS DMA (ASYNCcnt) when available.
// ---------------------------------------------------------------------------
__global__ __launch_bounds__(256) void gemm_bf16_kernel(
    const __bf16* __restrict__ A, const __bf16* __restrict__ W,
    const float* __restrict__ bias, float* __restrict__ C,
    int M, int N, int K) {
    __shared__ alignas(16) __bf16 As[128 * 40];   // stride 40 (80B) keeps b128 alignment
    __shared__ alignas(16) __bf16 Bs[64 * 40];

    const int tid  = threadIdx.x;
    const int lane = tid & 31, wave = tid >> 5;
    const int wm = wave >> 1, wn = wave & 1;      // 4 x 2 wave grid
    const int m0 = blockIdx.y * 128, n0 = blockIdx.x * 64;
    const int hf = lane >> 4, nn = lane & 15;

    v8f acc[2][2] = {};

    for (int kt = 0; kt < K; kt += 32) {
        __syncthreads();
        {   // A tile: 128 rows x 32 cols, 2 threads/row x 16 cols
            const int r = tid >> 1, c = (tid & 1) * 16;
            const __bf16* gp = A + (size_t)(m0 + r) * K + kt + c;
            __bf16* lp = &As[r * 40 + c];
#if HAVE_ASYNC_LDS
            async_cp32(gp, lp);
#else
            const uint4* g4 = reinterpret_cast<const uint4*>(gp);
            uint4* l4 = reinterpret_cast<uint4*>(lp);
            l4[0] = g4[0]; l4[1] = g4[1];
#endif
        }
        if (tid < 128) {  // W tile: 64 rows x 32 cols
            const int r = tid >> 1, c = (tid & 1) * 16;
            const __bf16* gp = W + (size_t)(n0 + r) * K + kt + c;
            __bf16* lp = &Bs[r * 40 + c];
#if HAVE_ASYNC_LDS
            async_cp32(gp, lp);
#else
            const uint4* g4 = reinterpret_cast<const uint4*>(gp);
            uint4* l4 = reinterpret_cast<uint4*>(lp);
            l4[0] = g4[0]; l4[1] = g4[1];
#endif
        }
#if HAVE_ASYNC_LDS
        __builtin_amdgcn_s_wait_asynccnt(0);
#endif
        __syncthreads();

        const v16bf a0 = frag_ld(&As[(wm * 32 + 0)  * 40], 40);
        const v16bf a1 = frag_ld(&As[(wm * 32 + 16) * 40], 40);
        const v16bf b0 = frag_ld(&Bs[(wn * 32 + 0)  * 40], 40);
        const v16bf b1 = frag_ld(&Bs[(wn * 32 + 16) * 40], 40);
        acc[0][0] = wmma_bf16(a0, b0, acc[0][0]);
        acc[0][1] = wmma_bf16(a0, b1, acc[0][1]);
        acc[1][0] = wmma_bf16(a1, b0, acc[1][0]);
        acc[1][1] = wmma_bf16(a1, b1, acc[1][1]);
    }

#pragma unroll
    for (int mt = 0; mt < 2; ++mt)
#pragma unroll
        for (int nt = 0; nt < 2; ++nt) {
            const int n = n0 + wn * 32 + nt * 16 + nn;
            const float bv = bias ? bias[n] : 0.f;
#pragma unroll
            for (int r = 0; r < 8; ++r) {
                const int m = m0 + wm * 32 + mt * 16 + r + hf * 8;
                C[(size_t)m * N + n] = acc[mt][nt][r] + bv;
            }
        }
}

// ---------------------------------------------------------------------------
// LSH bucketing: rotated = qk . rot ; bucket = argmax([r, -r]) ; +round*32
// one thread per (bh, round, pos); rotations cached in LDS.
// ---------------------------------------------------------------------------
__global__ __launch_bounds__(256) void bucket_kernel(
    const float* __restrict__ QK, const float* __restrict__ rot,
    int* __restrict__ bucketid) {
    __shared__ float rl[DD * NHASH * 16];  // [d][round][i], 8KB
    for (int i = threadIdx.x; i < DD * NHASH * 16; i += 256) rl[i] = rot[i];
    __syncthreads();

    const int id    = blockIdx.x * 256 + threadIdx.x;   // 48*8192 total, exact
    const int pos   = id & (TT - 1);
    const int round = (id >> 12) & 1;
    const int bh    = id >> 13;
    const int bb = bh / HH, hh = bh % HH;

    const float* q = QK + ((size_t)(bb * TT + pos)) * HIDN + hh * DD;
    float accv[16] = {};
#pragma unroll 4
    for (int d = 0; d < DD; ++d) {
        const float x = q[d];
        const float* rr = &rl[d * (NHASH * 16) + round * 16];
#pragma unroll
        for (int i = 0; i < 16; ++i) accv[i] += x * rr[i];
    }
    float bv = accv[0]; int bj = 0;
#pragma unroll
    for (int j = 1; j < 16; ++j) if (accv[j] > bv) { bv = accv[j]; bj = j; }
#pragma unroll
    for (int j = 0; j < 16; ++j) { const float v = -accv[j]; if (v > bv) { bv = v; bj = 16 + j; } }
    bucketid[(size_t)bh * TOTALN + round * TT + pos] = bj + round * NBUCK;
}

// ---------------------------------------------------------------------------
// Stable counting sort per bh: key = (bucket, pos, round). One wave per block.
// perm[bh][slot] = original index (round*T + pos), sorted order.
// ---------------------------------------------------------------------------
__global__ void sort_kernel(const int* __restrict__ bucketid, int* __restrict__ perm) {
    __shared__ unsigned hist[CBUCK];
    __shared__ unsigned cnt[CBUCK];
    __shared__ int bsh[32];
    const int lane = threadIdx.x;
    const int bh   = blockIdx.x;
    const int* bb = bucketid + (size_t)bh * TOTALN;

    hist[lane] = 0; hist[lane + 32] = 0;
    __syncthreads();
    for (int i = lane; i < TOTALN; i += 32) atomicAdd(&hist[bb[i]], 1u);
    __syncthreads();
    if (lane == 0) {
        unsigned run = 0;
        for (int j = 0; j < CBUCK; ++j) { const unsigned c = hist[j]; cnt[j] = run; run += c; }
    }
    __syncthreads();

    // scan items in (pos, round) order -> stable (bucket, pos, round) placement
    for (int g = 0; g < TOTALN / 32; ++g) {
        const int p = g * 32 + lane;
        const int pos = p >> 1, round = p & 1;
        const int idx = round * TT + pos;
        const int bmy = bb[idx];
        bsh[lane] = bmy;
        __syncthreads();
        int peers = 0;
        for (int j = 0; j < 32; ++j) if (j < lane && bsh[j] == bmy) ++peers;
        perm[(size_t)bh * TOTALN + cnt[bmy] + peers] = idx;
        __syncthreads();
        atomicAdd(&cnt[bmy], 1u);
        __syncthreads();
    }
}

// ---------------------------------------------------------------------------
// Chunked LSH attention, flash-style online softmax, WMMA inner products.
// Block = (bh, chunk): 128 queries vs 256 keys (chunk + previous chunk).
// Each of 8 waves owns 16 query rows. Outputs scattered to unsorted slots.
// ---------------------------------------------------------------------------
#define KS_STRIDE 72
#define VT_STRIDE 264
#define PB_STRIDE 40

__global__ __launch_bounds__(256) void lsh_attn_kernel(
    const float* __restrict__ QK, const float* __restrict__ Vf,
    const float* __restrict__ mask, const int* __restrict__ perm,
    float* __restrict__ o_un, float* __restrict__ lse_un) {
    extern __shared__ char smem[];
    int*    qposL = (int*)smem;                       // [128]
    int*    qoiL  = qposL + CL;                       // [128]
    int*    qmL   = qoiL + CL;                        // [128]
    int*    kposL = qmL + CL;                         // [256]
    int*    kmL   = kposL + 2 * CL;                   // [256]
    __bf16* ks    = (__bf16*)(kmL + 2 * CL);          // [256][72] normalized keys
    __bf16* vsT   = ks + 256 * KS_STRIDE;             // [64][264] v transposed
    __bf16* pb    = vsT + DD * VT_STRIDE;             // [8][16][40] P bounce

    const int chunk = blockIdx.x;
    const int bh    = blockIdx.y;
    const int bb = bh / HH, hh = bh % HH;
    const int prev  = (chunk + NCHUNK - 1) & (NCHUNK - 1);
    const int tid   = threadIdx.x;
    const int lane  = tid & 31, wave = tid >> 5;
    const int hf = lane >> 4, nn = lane & 15;
    const int* permb = perm + (size_t)bh * TOTALN;

    // ---- gather keys/values: one row per thread (256 rows) ----
    {
        const int j    = tid;
        const int slot = (j < CL) ? (chunk * CL + j) : (prev * CL + (j - CL));
        const int oi   = permb[slot];
        const int pos  = oi & (TT - 1);
        kposL[j] = pos;
        kmL[j]   = (mask[(size_t)bb * TT + pos] != 0.f) ? 1 : 0;
        const float* kr = QK + ((size_t)(bb * TT + pos)) * HIDN + hh * DD;
        float ss = 0.f;
#pragma unroll 8
        for (int d = 0; d < DD; ++d) { const float x = kr[d]; ss += x * x; }
        const float sc = 1.f / fmaxf(sqrtf(ss), 1e-12f);
#pragma unroll 8
        for (int d = 0; d < DD; ++d) ks[j * KS_STRIDE + d] = (__bf16)(kr[d] * sc);
        const float* vr = Vf + ((size_t)(bb * TT + pos)) * HIDN + hh * DD;
#pragma unroll 8
        for (int d = 0; d < DD; ++d) vsT[d * VT_STRIDE + j] = (__bf16)vr[d];
    }
    if (tid < CL) {
        const int oi  = permb[chunk * CL + tid];
        const int pos = oi & (TT - 1);
        qoiL[tid] = oi; qposL[tid] = pos;
        qmL[tid]  = (mask[(size_t)bb * TT + pos] != 0.f) ? 1 : 0;
    }
    __syncthreads();

    // ---- q fragments straight from global (unnormalized) ----
    const int qrow = wave * 16 + nn;
    const float* qg = QK + ((size_t)(bb * TT + qposL[qrow])) * HIDN + hh * DD;
    const v16bf qa0 = frag_row_f32(qg);
    const v16bf qa1 = frag_row_f32(qg + 32);

    float mrow[8], srow[8];
    int qp[8], qm[8];
    v8f o[4] = {};
#pragma unroll
    for (int r = 0; r < 8; ++r) {
        const int gq = wave * 16 + r + hf * 8;
        qp[r] = qposL[gq]; qm[r] = qmL[gq];
        mrow[r] = -INFINITY; srow[r] = 0.f;
    }

    const float SCALE = 0.125f;  // 64^-0.5
    __bf16* pbw = pb + wave * 16 * PB_STRIDE;

    for (int kb = 0; kb < 8; ++kb) {
        const int k0 = kb * 32;
        // S = q @ k^T for 32 keys (two 16-col D tiles, K=64 = two wmma each)
        v8f s0 = {}, s1 = {};
        s0 = wmma_bf16(qa0, frag_ld(&ks[(size_t)k0 * KS_STRIDE], KS_STRIDE), s0);
        s0 = wmma_bf16(qa1, frag_ld(&ks[(size_t)k0 * KS_STRIDE + 32], KS_STRIDE), s0);
        s1 = wmma_bf16(qa0, frag_ld(&ks[(size_t)(k0 + 16) * KS_STRIDE], KS_STRIDE), s1);
        s1 = wmma_bf16(qa1, frag_ld(&ks[(size_t)(k0 + 16) * KS_STRIDE + 32], KS_STRIDE), s1);

        const int kp0 = kposL[k0 + nn],      km0 = kmL[k0 + nn];
        const int kp1 = kposL[k0 + 16 + nn], km1 = kmL[k0 + 16 + nn];

#pragma unroll
        for (int r = 0; r < 8; ++r) {
            float v0 = s0[r] * SCALE, v1 = s1[r] * SCALE;
            if (!(qm[r] && km0)) v0 = -1e9f;
            if (!(qm[r] && km1)) v1 = -1e9f;
            if (qp[r] == kp0) v0 = -5e4f;        // self-attn overrides pad mask
            if (qp[r] == kp1) v1 = -5e4f;
            float rm = fmaxf(v0, v1);
#pragma unroll
            for (int s = 1; s < 16; s <<= 1) rm = fmaxf(rm, __shfl_xor(rm, s, 32));
            const float mn = fmaxf(mrow[r], rm);
            const float fr = __expf(mrow[r] - mn);
            const float p0 = __expf(v0 - mn), p1 = __expf(v1 - mn);
            float ps = p0 + p1;
#pragma unroll
            for (int s = 1; s < 16; s <<= 1) ps += __shfl_xor(ps, s, 32);
            srow[r] = srow[r] * fr + ps;
            mrow[r] = mn;
#pragma unroll
            for (int j = 0; j < 4; ++j) o[j][r] *= fr;
            const int rm16 = r + hf * 8;
            pbw[rm16 * PB_STRIDE + nn]      = (__bf16)p0;   // D layout -> row-major
            pbw[rm16 * PB_STRIDE + 16 + nn] = (__bf16)p1;
        }
        __builtin_amdgcn_wave_barrier();
        const v16bf pa = frag_ld(pbw, PB_STRIDE);           // P as A fragment (16x32)
#pragma unroll
        for (int j = 0; j < 4; ++j) {
            const v16bf bv = frag_ld(&vsT[(size_t)(j * 16) * VT_STRIDE + k0], VT_STRIDE);
            o[j] = wmma_bf16(pa, bv, o[j]);
        }
        __builtin_amdgcn_wave_barrier();
    }

    // ---- normalize, scatter to unsorted (round,pos) slots, store LSE ----
#pragma unroll
    for (int r = 0; r < 8; ++r) {
        const int gq = wave * 16 + r + hf * 8;
        const int oi = qoiL[gq];
        const float inv = 1.f / srow[r];
        const size_t ob = ((size_t)bh * TOTALN + oi) * DD;
#pragma unroll
        for (int j = 0; j < 4; ++j) o_un[ob + j * 16 + nn] = o[j][r] * inv;
        if (nn == 0) lse_un[(size_t)bh * TOTALN + oi] = mrow[r] + __logf(srow[r]);
    }
}

// ---------------------------------------------------------------------------
// Combine hash rounds (softmax over per-round LSE), emit merged-heads bf16.
// One wave per (bh, pos) row.
// ---------------------------------------------------------------------------
__global__ __launch_bounds__(256) void combine_kernel(
    const float* __restrict__ o_un, const float* __restrict__ lse_un,
    __bf16* __restrict__ attnb) {
    const int row  = blockIdx.x * 8 + (threadIdx.x >> 5);
    const int lane = threadIdx.x & 31;
    const int bh = row >> 12, pos = row & (TT - 1);
    const int bb = bh / HH, hh = bh % HH;

    const float l0 = lse_un[(size_t)bh * TOTALN + pos];
    const float l1 = lse_un[(size_t)bh * TOTALN + TT + pos];
    const float mx = fmaxf(l0, l1);
    float w0 = __expf(l0 - mx), w1 = __expf(l1 - mx);
    const float inv = 1.f / (w0 + w1);
    w0 *= inv; w1 *= inv;

    const size_t base0 = ((size_t)bh * TOTALN + pos) * DD;
    const size_t base1 = ((size_t)bh * TOTALN + TT + pos) * DD;
    const size_t ob    = ((size_t)(bb * TT + pos)) * HIDN + hh * DD;
#pragma unroll
    for (int c = lane; c < DD; c += 32)
        attnb[ob + c] = (__bf16)(w0 * o_un[base0 + c] + w1 * o_un[base1 + c]);
}

// ---------------------------------------------------------------------------
// host launcher
// ---------------------------------------------------------------------------
extern "C" void kernel_launch(void* const* d_in, const int* in_sizes, int n_in,
                              void* d_out, int out_size, void* d_ws, size_t ws_size,
                              hipStream_t stream) {
    (void)in_sizes; (void)n_in; (void)out_size; (void)ws_size;

    const float* X     = (const float*)d_in[0];
    const float* mask  = (const float*)d_in[1];
    const float* W_qk  = (const float*)d_in[2];
    const float* W_v   = (const float*)d_in[3];
    const float* rot   = (const float*)d_in[4];
    const float* W_to  = (const float*)d_in[5];
    const float* b_to  = (const float*)d_in[6];
    const float* W_o   = (const float*)d_in[7];
    const float* b_o   = (const float*)d_in[8];
    float* out = (float*)d_out;

    // workspace carve
    char* w = (char*)d_ws;
    size_t off = 0;
    auto carve = [&](size_t bytes) { char* p = w + off; off += (bytes + 255) & ~(size_t)255; return p; };

    const size_t MK = (size_t)MTOT * HIDN;          // 16384*768
    const size_t WW = (size_t)HIDN * HIDN;          // 768*768
    __bf16* Xb    = (__bf16*)carve(MK * 2);
    __bf16* Wqkb  = (__bf16*)carve(WW * 2);
    __bf16* Wvb   = (__bf16*)carve(WW * 2);
    __bf16* Wtob  = (__bf16*)carve(WW * 2);
    __bf16* Wob   = (__bf16*)carve(WW * 2);
    float*  QKf   = (float*)carve(MK * 4);
    float*  Vff   = (float*)carve(MK * 4);
    int*    bucketid = (int*)carve((size_t)BHN * TOTALN * 4);
    int*    perm     = (int*)carve((size_t)BHN * TOTALN * 4);
    float*  o_un     = (float*)carve((size_t)BHN * TOTALN * DD * 4);
    float*  lse_un   = (float*)carve((size_t)BHN * TOTALN * 4);
    __bf16* attnb    = (__bf16*)carve(MK * 2);
    float*  tmpf     = (float*)carve(MK * 4);
    __bf16* tmpb     = (__bf16*)carve(MK * 2);

    // 1) bf16 conversions
    f32_to_bf16_kernel<<<(int)((MK + 255) / 256), 256, 0, stream>>>(X, Xb, MK);
    f32_to_bf16_kernel<<<(int)((WW + 255) / 256), 256, 0, stream>>>(W_qk, Wqkb, WW);
    f32_to_bf16_kernel<<<(int)((WW + 255) / 256), 256, 0, stream>>>(W_v, Wvb, WW);
    f32_to_bf16_kernel<<<(int)((WW + 255) / 256), 256, 0, stream>>>(W_to, Wtob, WW);
    f32_to_bf16_kernel<<<(int)((WW + 255) / 256), 256, 0, stream>>>(W_o, Wob, WW);

    const dim3 gg(HIDN / 64, MTOT / 128);  // (12, 128)
    // 2) qk / v projections
    gemm_bf16_kernel<<<gg, 256, 0, stream>>>(Xb, Wqkb, nullptr, QKf, MTOT, HIDN, HIDN);
    gemm_bf16_kernel<<<gg, 256, 0, stream>>>(Xb, Wvb,  nullptr, Vff, MTOT, HIDN, HIDN);

    // 3) LSH buckets + stable counting sort
    bucket_kernel<<<(BHN * TOTALN) / 256, 256, 0, stream>>>(QKf, rot, bucketid);
    sort_kernel<<<BHN, 32, 0, stream>>>(bucketid, perm);

    // 4) chunked attention (dynamic LDS)
    const size_t smem = (size_t)(3 * CL + 2 * 2 * CL) * 4          // pos/oi/mask arrays
                      + (size_t)256 * KS_STRIDE * 2                // ks
                      + (size_t)DD * VT_STRIDE * 2                 // vsT
                      + (size_t)8 * 16 * PB_STRIDE * 2;            // P bounce
    (void)hipFuncSetAttribute(reinterpret_cast<const void*>(lsh_attn_kernel),
                              hipFuncAttributeMaxDynamicSharedMemorySize, (int)smem);
    lsh_attn_kernel<<<dim3(NCHUNK, BHN), 256, smem, stream>>>(QKf, Vff, mask, perm, o_un, lse_un);

    // 5) combine rounds -> merged-heads bf16
    combine_kernel<<<(BHN * TT) / 8, 256, 0, stream>>>(o_un, lse_un, attnb);

    // 6) output projections
    gemm_bf16_kernel<<<gg, 256, 0, stream>>>(attnb, Wtob, b_to, tmpf, MTOT, HIDN, HIDN);
    f32_to_bf16_kernel<<<(int)((MK + 255) / 256), 256, 0, stream>>>(tmpf, tmpb, MK);
    gemm_bf16_kernel<<<gg, 256, 0, stream>>>(tmpb, Wob, b_o, out, MTOT, HIDN, HIDN);
}
